// EchoSpike_35957466202607
// MI455X (gfx1250) — matrix operand order repaired and checked by reading, not compiled, not run on hardware
//
#include <hip/hip_runtime.h>

#define T_STEPS 100
#define BATCH   128
#define NIN     784
#define NINP    832          // NIN padded to multiple of 64 (removes all GEMM edge guards)
#define H1      1024
#define H2      1024
#define BT      (T_STEPS*BATCH)   // 12800
#define BETA    0.95f
#define CY0     1e-4f
#define PI_F    3.14159265358979323846f

typedef __attribute__((ext_vector_type(16))) _Float16 v16h;
typedef __attribute__((ext_vector_type(8)))  _Float16 v8h;
typedef __attribute__((ext_vector_type(8)))  float    v8f;

// ---------------- WMMA fragment helpers (gfx1250 16x16x32 f16 layouts) ----------------
// A (16x32, row-major source, lda in halves):
//   lane l<16 : M=l,    halves 0..7 -> K=0..7,  halves 8..15 -> K=16..23
//   lane l>=16: M=l-16, halves 0..7 -> K=8..15, halves 8..15 -> K=24..31
__device__ __forceinline__ v16h frag_a(const _Float16* __restrict__ A, int lda,
                                       int m0, int k0, int lane) {
  const int l = lane & 15, hi = lane >> 4;
  const _Float16* p = A + (size_t)(m0 + l) * lda + (k0 + hi * 8);
  v8h lo  = *(const v8h*)(p);
  v8h hi8 = *(const v8h*)(p + 16);
  v16h a;
#pragma unroll
  for (int j = 0; j < 8; ++j) { a[j] = lo[j]; a[8 + j] = hi8[j]; }
  return a;
}

// B (32x16, row-major source, ldb in halves): lane l holds column N=n0+l%16, K in same order as frag_a.
__device__ __forceinline__ v16h frag_b(const _Float16* __restrict__ Bm, int ldb,
                                       int k0, int n0, int lane) {
  const int l = lane & 15, hi = lane >> 4;
  const _Float16* p = Bm + (size_t)(k0 + hi * 8) * ldb + n0 + l;
  v16h b;
#pragma unroll
  for (int j = 0; j < 8; ++j) { b[j] = p[(size_t)j * ldb]; b[8 + j] = p[(size_t)(16 + j) * ldb]; }
  return b;
}

// ---------------- Wave-tile GEMM: C(MxN,f32) = sgn * A(MxK,f16) @ B(KxN,f16) ----------
// Requirements (guaranteed by caller): M%16==0, N%64==0, K%32==0,
// grid covers exactly (M/16)*(N/64) waves. Straight-line inner loop, no EXEC games.
// SCALED: multiply B row k by bscale[k & 127] (factor per B element, by its k; folds the dL gate).
template <bool SCALED>
__global__ __launch_bounds__(128)
void gemm16x64(const _Float16* __restrict__ A, int lda,
               const _Float16* __restrict__ Bm, int ldb,
               float* __restrict__ C, int ldc,
               int N, int K,
               const float* __restrict__ bscale, float sgn) {
  const int lane   = threadIdx.x & 31;
  const int wave   = (blockIdx.x * blockDim.x + threadIdx.x) >> 5;
  const int tn4cnt = N >> 6;
  const int tm = wave / tn4cnt;
  const int tn = wave - tm * tn4cnt;
  const int m0 = tm * 16, n0 = tn * 64;

  v8f acc[4];
#pragma unroll
  for (int j = 0; j < 4; ++j) acc[j] = (v8f){0.f,0.f,0.f,0.f,0.f,0.f,0.f,0.f};

  for (int k0 = 0; k0 < K; k0 += 32) {
    v16h a = frag_a(A, lda, m0, k0, lane);
    _Float16 sc = (_Float16)1.0f;
    if (SCALED) sc = (_Float16)bscale[(k0 + lane) & (BATCH - 1)]; // k = t*128+b -> b = k&127
#pragma unroll
    for (int j = 0; j < 4; ++j) {
      v16h b = frag_b(Bm, ldb, k0, n0 + 16 * j, lane);
      if (SCALED) {
#pragma unroll
        for (int e = 0; e < 16; ++e) b[e] *= (_Float16)bscale[(k0 + (lane >> 4) * 8 + (e & 7) + (e >> 3) * 16) & (BATCH - 1)];   // v_pk_mul_f16, gate in {0,1}
      }
      acc[j] = __builtin_amdgcn_wmma_f32_16x16x32_f16(
          false, a, false, b, (short)0, acc[j], false, false);
    }
  }
  // C/D layout: VGPR r, lane l<16 -> (M=m0+r, N=n0+l); lane>=16 -> (M=m0+8+r, N=n0+l-16)
  const int l = lane & 15, hi = lane >> 4;
#pragma unroll
  for (int j = 0; j < 4; ++j) {
#pragma unroll
    for (int r = 0; r < 8; ++r)
      C[(size_t)(m0 + r + hi * 8) * ldc + (n0 + 16 * j) + l] = sgn * acc[j][r];
  }
}

// ---------------- Elementwise / setup kernels ----------------
__global__ void zero_f32(float* __restrict__ p, int n) {
  int i = blockIdx.x * 256 + threadIdx.x;
  if (i < n) p[i] = 0.f;
}

__global__ void copy_f32(const float* __restrict__ s, float* __restrict__ d, int n) {
  int i = blockIdx.x * 256 + threadIdx.x;
  if (i < n) d[i] = s[i];
}

// d[r, 0:NIN] = s[r, 0:NIN] with source pitch NINP (strip the pad columns of gW1).
__global__ void copy_slice(const float* __restrict__ s, float* __restrict__ d, int rows) {
  int i = blockIdx.x * 256 + threadIdx.x;
  if (i >= rows * NIN) return;
  int r = i / NIN, c = i - r * NIN;
  d[i] = s[(size_t)r * NINP + c];
}

// WT[k][n] = (k<cols) ? (f16)W[n][k] : 0   — LDS tiled transpose, coalesced both sides.
__global__ void transpose_to_f16(const float* __restrict__ W, _Float16* __restrict__ WT,
                                 int rows, int cols) {
  __shared__ float sh[16][17];
  const int k0 = blockIdx.x * 16, n0 = blockIdx.y * 16;
  const int tx = threadIdx.x, ty = threadIdx.y;
  const int k = k0 + tx;
  sh[ty][tx] = (k < cols) ? W[(size_t)(n0 + ty) * cols + k] : 0.f;
  __syncthreads();
  WT[(size_t)(k0 + ty) * rows + (n0 + tx)] = (_Float16)sh[tx][ty];
}

// Per step: it1 = beta*it1 + x ; emit f16 x (padded) and f16 it1 trace row (padded).
__global__ void trace1_kernel(const float* __restrict__ inp, float* __restrict__ it1s,
                              _Float16* __restrict__ xcur, _Float16* __restrict__ it1buf,
                              int t) {
  const int c = blockIdx.x * 256 + threadIdx.x;
  const int b = blockIdx.y;
  if (c >= NINP) return;
  const size_t row = (size_t)t * BATCH + b;
  if (c < NIN) {
    float x  = inp[row * NIN + c];
    float it = BETA * it1s[b * NIN + c] + x;
    it1s[b * NIN + c] = it;
    xcur[b * NINP + c]     = (_Float16)x;
    it1buf[row * NINP + c] = (_Float16)it;
  } else {
    xcur[b * NINP + c]     = (_Float16)0.f;
    it1buf[row * NINP + c] = (_Float16)0.f;
  }
}

// LIF layer 1: mem update, spikes (f16 for next GEMM), eligibility a1 -> transposed (H1 x BT)
// stored via LDS tile transpose, layer-2 input trace it2, deterministic sample-loss partials.
__global__ void lif1_kernel(const float* __restrict__ cur1, float* __restrict__ m1,
                            const float* __restrict__ prev1, const int* __restrict__ bf,
                            _Float16* __restrict__ s1cur, _Float16* __restrict__ a1T,
                            float* __restrict__ it2s, _Float16* __restrict__ it2buf,
                            float* __restrict__ slacc, int t) {
  __shared__ _Float16 shA[16][17];
  __shared__ float    shR[16][17];
  const int hx = threadIdx.x, by = threadIdx.y;
  const int h = blockIdx.x * 16 + hx;
  const int b = blockIdx.y * 16 + by;
  const int idx = b * H1 + h;
  const float bff = (float)bf[0];

  float m = m1[idx];
  float reset = (m > 1.f) ? 1.f : 0.f;            // reset from incoming mem
  m = BETA * m + cur1[idx] - reset;
  m1[idx] = m;
  float s = (m > 1.f) ? 1.f : 0.f;
  s1cur[idx] = (_Float16)s;

  float p = prev1[idx];
  float d = PI_F * (m - 1.f);
  float sur = 1.f / (PI_F * (1.f + d * d));
  shA[hx][by] = (_Float16)(bff * p * sur);        // bf folded into eligibility
  shR[by][hx] = s * p;

  const size_t row = (size_t)t * BATCH + b;
  float it = BETA * it2s[idx] + s;
  it2s[idx] = it;
  it2buf[row * H1 + h] = (_Float16)it;
  __syncthreads();

  { // transposed write: consecutive lanes -> consecutive batch index
    const int hh = blockIdx.x * 16 + by;
    const int bb = blockIdx.y * 16 + hx;
    a1T[(size_t)hh * BT + (size_t)t * BATCH + bb] = shA[by][hx];
  }
  for (int o = 8; o > 0; o >>= 1) {
    if (hx < o) shR[by][hx] += shR[by][hx + o];
    __syncthreads();
  }
  if (hx == 0) slacc[b * (H1 / 16) + blockIdx.x] += -bff * shR[by][0]; // unique owner: deterministic
}

// LIF layer 2: also writes spk_seq output and running st2.
__global__ void lif2_kernel(const float* __restrict__ cur2, float* __restrict__ m2,
                            const float* __restrict__ prev2, const int* __restrict__ bf,
                            float* __restrict__ out_spk, float* __restrict__ st2,
                            _Float16* __restrict__ a2T, float* __restrict__ slacc, int t) {
  __shared__ _Float16 shA[16][17];
  __shared__ float    shR[16][17];
  const int hx = threadIdx.x, by = threadIdx.y;
  const int h = blockIdx.x * 16 + hx;
  const int b = blockIdx.y * 16 + by;
  const int idx = b * H2 + h;
  const float bff = (float)bf[0];

  float m = m2[idx];
  float reset = (m > 1.f) ? 1.f : 0.f;
  m = BETA * m + cur2[idx] - reset;
  m2[idx] = m;
  float s = (m > 1.f) ? 1.f : 0.f;

  const size_t row = (size_t)t * BATCH + b;
  out_spk[row * H2 + h] = s;
  st2[idx] += s * (1.f / (float)T_STEPS);

  float p = prev2[idx];
  float d = PI_F * (m - 1.f);
  float sur = 1.f / (PI_F * (1.f + d * d));
  shA[hx][by] = (_Float16)(bff * p * sur);
  shR[by][hx] = s * p;
  __syncthreads();

  {
    const int hh = blockIdx.x * 16 + by;
    const int bb = blockIdx.y * 16 + hx;
    a2T[(size_t)hh * BT + (size_t)t * BATCH + bb] = shA[by][hx];
  }
  for (int o = 8; o > 0; o >>= 1) {
    if (hx < o) shR[by][hx] += shR[by][hx + o];
    __syncthreads();
  }
  if (hx == 0) slacc[b * (H2 / 16) + blockIdx.x] += -bff * shR[by][0];
}

// dL gates from accumulated sample losses.
__global__ void finalize_dl(const float* __restrict__ sl1acc, const float* __restrict__ sl2acc,
                            float* __restrict__ dL1, float* __restrict__ dL2) {
  const int tid = threadIdx.x;
  if (tid < BATCH) {
    float s = 0.f;
    for (int j = 0; j < H1 / 16; ++j) s += sl1acc[tid * (H1 / 16) + j];
    dL1[tid] = ((s + CY0 * T_STEPS * H1) > 0.f) ? 1.f : 0.f;
  } else if (tid < 2 * BATCH) {
    const int b = tid - BATCH;
    float s = 0.f;
    for (int j = 0; j < H2 / 16; ++j) s += sl2acc[b * (H2 / 16) + j];
    dL2[b] = ((s + CY0 * T_STEPS * H2) > 0.f) ? 1.f : 0.f;
  }
}

// ---------------- Host driver ----------------
extern "C" void kernel_launch(void* const* d_in, const int* in_sizes, int n_in,
                              void* d_out, int out_size, void* d_ws, size_t ws_size,
                              hipStream_t stream) {
  const float* inp   = (const float*)d_in[0];   // (T,B,NIN) {0,1}
  const float* W1    = (const float*)d_in[1];   // (H1,NIN)
  const float* W2    = (const float*)d_in[2];   // (H2,H1)
  const float* prev1 = (const float*)d_in[3];   // (B,H1)
  const float* prev2 = (const float*)d_in[4];   // (B,H2)
  const int*   bf    = (const int*)d_in[5];

  float* out      = (float*)d_out;
  float* out_spk  = out;                                     // (T,B,H2)
  float* out_st2  = out_spk + (size_t)T_STEPS * BATCH * H2;  // (B,H2)
  float* out_gW1  = out_st2 + (size_t)BATCH * H2;            // (H1,NIN)
  float* out_gW2  = out_gW1 + (size_t)H1 * NIN;              // (H2,H1)

  char* ws = (char*)d_ws;
  size_t off = 0;
  auto alloc = [&](size_t bytes) -> char* {
    char* p = ws + off;
    off = (off + bytes + 255) & ~(size_t)255;
    return p;
  };

  // contiguous zero-initialized f32 state
  const size_t zcount = (size_t)BATCH * H1 + (size_t)BATCH * H2 + (size_t)BATCH * NIN +
                        (size_t)BATCH * H1 + (size_t)BATCH * H2 +
                        (size_t)BATCH * (H1 / 16) + (size_t)BATCH * (H2 / 16);
  float* zbase  = (float*)alloc(zcount * 4);
  float* m1     = zbase;
  float* m2     = m1 + (size_t)BATCH * H1;
  float* it1s   = m2 + (size_t)BATCH * H2;
  float* it2s   = it1s + (size_t)BATCH * NIN;
  float* st2    = it2s + (size_t)BATCH * H1;
  float* sl1acc = st2 + (size_t)BATCH * H2;
  float* sl2acc = sl1acc + (size_t)BATCH * (H1 / 16);

  _Float16* W1T    = (_Float16*)alloc((size_t)NINP * H1 * 2);
  _Float16* W2T    = (_Float16*)alloc((size_t)H1 * H2 * 2);
  _Float16* xcur   = (_Float16*)alloc((size_t)BATCH * NINP * 2);
  _Float16* s1cur  = (_Float16*)alloc((size_t)BATCH * H1 * 2);
  float*    cur1   = (float*)alloc((size_t)BATCH * H1 * 4);
  float*    cur2   = (float*)alloc((size_t)BATCH * H2 * 4);
  float*    dL1    = (float*)alloc(BATCH * 4);
  float*    dL2    = (float*)alloc(BATCH * 4);
  float*    gW1p   = (float*)alloc((size_t)H1 * NINP * 4);   // padded gW1 staging
  _Float16* a1T    = (_Float16*)alloc((size_t)H1 * BT * 2);
  _Float16* a2T    = (_Float16*)alloc((size_t)H2 * BT * 2);
  _Float16* it1buf = (_Float16*)alloc((size_t)BT * NINP * 2);
  _Float16* it2buf = (_Float16*)alloc((size_t)BT * H1 * 2);

  zero_f32<<<(int)((zcount + 255) / 256), 256, 0, stream>>>(zbase, (int)zcount);
  transpose_to_f16<<<dim3(NINP / 16, H1 / 16), dim3(16, 16), 0, stream>>>(W1, W1T, H1, NIN);
  transpose_to_f16<<<dim3(H1 / 16, H2 / 16), dim3(16, 16), 0, stream>>>(W2, W2T, H2, H1);

  for (int t = 0; t < T_STEPS; ++t) {
    trace1_kernel<<<dim3((NINP + 255) / 256, BATCH), 256, 0, stream>>>(inp, it1s, xcur, it1buf, t);
    { // cur1 = xcur(128x832) @ W1T(832x1024) : 128 waves, 32 blocks
      const int waves = (BATCH / 16) * (H1 / 64);
      gemm16x64<false><<<waves / 4, 128, 0, stream>>>(
          xcur, NINP, W1T, H1, cur1, H1, H1, NINP, nullptr, 1.f);
    }
    lif1_kernel<<<dim3(H1 / 16, BATCH / 16), dim3(16, 16), 0, stream>>>(
        cur1, m1, prev1, bf, s1cur, a1T, it2s, it2buf, sl1acc, t);
    { // cur2 = s1(128x1024) @ W2T(1024x1024)
      const int waves = (BATCH / 16) * (H2 / 64);
      gemm16x64<false><<<waves / 4, 128, 0, stream>>>(
          s1cur, H1, W2T, H2, cur2, H2, H2, H1, nullptr, 1.f);
    }
    lif2_kernel<<<dim3(H2 / 16, BATCH / 16), dim3(16, 16), 0, stream>>>(
        cur2, m2, prev2, bf, out_spk, st2, a2T, sl2acc, t);
  }

  finalize_dl<<<1, 256, 0, stream>>>(sl1acc, sl2acc, dL1, dL2);

  { // gW1p = -(dL-gated) a1T(1024x12800) @ it1buf(12800x832) : 832 waves, 208 blocks
    const int waves = (H1 / 16) * (NINP / 64);
    gemm16x64<true><<<waves / 4, 128, 0, stream>>>(
        a1T, BT, it1buf, NINP, gW1p, NINP, NINP, BT, dL1, -1.f);
  }
  { // gW2 = -(dL-gated) a2T(1024x12800) @ it2buf(12800x1024) : 1024 waves, 256 blocks
    const int waves = (H2 / 16) * (H1 / 64);
    gemm16x64<true><<<waves / 4, 128, 0, stream>>>(
        a2T, BT, it2buf, H1, out_gW2, H1, H1, BT, dL2, -1.f);
  }

  copy_slice<<<(H1 * NIN + 255) / 256, 256, 0, stream>>>(gW1p, out_gW1, H1);
  copy_f32<<<(BATCH * H2 + 255) / 256, 256, 0, stream>>>(st2, out_st2, BATCH * H2);
}
